// AttentionMultiItems_85607288143990
// MI455X (gfx1250) — matrix operand order repaired and checked by reading, not compiled, not run on hardware
//
#include <hip/hip_runtime.h>

// MI455X / gfx1250, wave32. DIN attention: B=128, N=16, T=200, H=64.
// Compute-bound (~19.4 GFLOP vs ~7.5MB traffic) -> WMMA bf16 for the MLP,
// WMMA f32 16x16x4 for the exact output reduction.

#define B_ 128
#define N_ 16
#define T_ 200
#define H_ 64
#define NEGC (-4294967296.0f)

typedef __attribute__((ext_vector_type(16))) __bf16 v16bf;
typedef __attribute__((ext_vector_type(8)))  float  v8f;
typedef __attribute__((ext_vector_type(2)))  float  v2f;

struct BF32 { uint4 lo, hi; };   // 32B register image of a 16-elem bf16 fragment

__device__ __forceinline__ unsigned int pack2bf(float a, float b) {
  unsigned short ua = __builtin_bit_cast(unsigned short, (__bf16)a);
  unsigned short ub = __builtin_bit_cast(unsigned short, (__bf16)b);
  return ((unsigned int)ub << 16) | (unsigned int)ua;
}

__device__ __forceinline__ float sigmoidf_(float x) {
  return 1.0f / (1.0f + __expf(-x));
}

// ---------------------------------------------------------------------------
// Kernel 0: pack W1 (256x80) and W2 (padded 96x48) into B-fragment images.
// B-frag layout (16-bit, KxN=32x16): lane L holds col N=L%16; VGPR v packs
// rows K = 32*ki + 8*(L/16) + 2v + (v>=4 ? 8 : 0) + {0, 1}.  W1p: [ki<8][nt<5][lane][v] u32.
// W2p follows at u32 offset 10240: [ki<3][nt<3][lane][v], zero-padded.
// ---------------------------------------------------------------------------
__global__ void pack_weights_kernel(const float* __restrict__ W1,
                                    const float* __restrict__ W2,
                                    unsigned int* __restrict__ ws) {
  int i = blockIdx.x * 256 + threadIdx.x;
  if (i < 10240) {
    int v = i & 7, lane = (i >> 3) & 31, g = i >> 8;
    int nt = g % 5, ki = g / 5;
    int half = lane >> 4;
    int row0 = 32 * ki + 8 * half + 2 * v + (v >= 4 ? 8 : 0);
    int col  = 16 * nt + (lane & 15);
    ws[i] = pack2bf(W1[row0 * 80 + col], W1[(row0 + 1) * 80 + col]);
  } else if (i < 10240 + 2304) {
    int j = i - 10240;
    int v = j & 7, lane = (j >> 3) & 31, g = j >> 8;
    int nt = g % 3, ki = g / 3;
    int half = lane >> 4;
    int row0 = 32 * ki + 8 * half + 2 * v + (v >= 4 ? 8 : 0);
    int col  = 16 * nt + (lane & 15);
    float f0 = (row0     < 80 && col < 40) ? W2[row0 * 40 + col]       : 0.0f;
    float f1 = (row0 + 1 < 80 && col < 40) ? W2[(row0 + 1) * 40 + col] : 0.0f;
    ws[i] = pack2bf(f0, f1);
  }
}

// ---------------------------------------------------------------------------
// Kernel 1: main. grid = B*2; block = 256 (8 waves). WG (2b+j) owns rows
// n = 8j..8j+7 of batch b; wave w owns n = 8j+w (MLP over 13 t-tiles +
// its softmax). Waves 0-3 then do the 8x200 @ 200x64 output GEMM in f32 WMMA.
// ---------------------------------------------------------------------------
__launch_bounds__(256, 1)
__global__ void din_attention_kernel(const float* __restrict__ queries,
                                     const float* __restrict__ keys,
                                     const int*   __restrict__ keys_length,
                                     const float* __restrict__ b1,
                                     const float* __restrict__ b2,
                                     const float* __restrict__ W3,
                                     const float* __restrict__ b3,
                                     const unsigned int* __restrict__ ws,
                                     float* __restrict__ out) {
  __shared__ float qs[8][64];                 // q rows for the 8 n's
  __shared__ float sc[8][208];                // scores -> softmax weights
  __shared__ float b1s[80];
  __shared__ float b2s[48];
  __shared__ float w3s[40];
  __shared__ float b3s_sh;
  __shared__ int   len_sh;
  __shared__ __align__(16) unsigned short xbuf[8][16 * 96]; // per-wave X1/X2

  const int tid  = threadIdx.x;
  const int lane = tid & 31;
  const int wave = tid >> 5;
  const int half = lane >> 4;      // K-half selector in fragments
  const int lrow = lane & 15;      // M (A/C) or N (B/C) index

  const int b  = blockIdx.x >> 1;
  const int nb = (blockIdx.x & 1) * 8;

  for (int i = tid; i < 512; i += 256)
    qs[i >> 6][i & 63] = queries[(b * N_ + nb + (i >> 6)) * H_ + (i & 63)];
  if (tid < 80) b1s[tid] = b1[tid];
  if (tid < 48) b2s[tid] = (tid < 40) ? b2[tid] : 0.0f;
  if (tid < 40) w3s[tid] = W3[tid];
  if (tid == 0) { b3s_sh = b3[0]; len_sh = keys_length[b]; }
  __syncthreads();

  const float* qrow = &qs[wave][0];
  unsigned short* x1 = &xbuf[wave][0];        // 16x96 bf16
  float* xf = (float*)x1;                     // reused as 16x48 f32
  const uint4* W1p = (const uint4*)ws;
  const uint4* W2p = (const uint4*)(ws + 10240);

  for (int tt = 0; tt < 13; ++tt) {
    const int t0 = tt * 16;
    const int tk = min(t0 + lrow, T_ - 1);    // clamp tail rows (masked later)
    const float* krow = keys + (b * T_ + tk) * H_;

    // ---- GEMM1: feat(16x256 bf16) @ W1(256x80 bf16) -> 16x80 f32 ----
    v8f acc[5];
    #pragma unroll
    for (int nt = 0; nt < 5; ++nt) acc[nt] = (v8f){0,0,0,0,0,0,0,0};

    #pragma unroll
    for (int ki = 0; ki < 8; ++ki) {
      // A-frag: lane holds row M=lrow; elements 0..7  = cols 32ki+8*half+e,
      //                                elements 8..15 = cols 32ki+16+8*half+e.
      // Both 8-runs fall inside section sec = ki>>1 (0:q 1:k 2:q-k 3:q*k).
      const int sec = ki >> 1;
      const int hb  = 32 * (ki & 1) + 8 * half;   // h-offset inside section
      __bf16 tmp[16];
      #pragma unroll
      for (int run = 0; run < 2; ++run) {
        const int h = hb + 16 * run;
        float qv[8] = {}, kv[8] = {};
        if (sec != 1) {
          float4 a0 = *(const float4*)(qrow + h);
          float4 a1 = *(const float4*)(qrow + h + 4);
          qv[0]=a0.x; qv[1]=a0.y; qv[2]=a0.z; qv[3]=a0.w;
          qv[4]=a1.x; qv[5]=a1.y; qv[6]=a1.z; qv[7]=a1.w;
        }
        if (sec != 0) {
          float4 c0 = *(const float4*)(krow + h);
          float4 c1 = *(const float4*)(krow + h + 4);
          kv[0]=c0.x; kv[1]=c0.y; kv[2]=c0.z; kv[3]=c0.w;
          kv[4]=c1.x; kv[5]=c1.y; kv[6]=c1.z; kv[7]=c1.w;
        }
        #pragma unroll
        for (int e = 0; e < 8; ++e) {
          float f = (sec == 0) ? qv[e]
                  : (sec == 1) ? kv[e]
                  : (sec == 2) ? (qv[e] - kv[e])
                               : (qv[e] * kv[e]);
          tmp[run * 8 + e] = (__bf16)f;
        }
      }
      v16bf afrag;
      #pragma unroll
      for (int e = 0; e < 16; ++e) afrag[e] = tmp[e];

      #pragma unroll
      for (int nt = 0; nt < 5; ++nt) {
        const uint4* p = W1p + (((ki * 5 + nt) * 32 + lane) << 1);
        BF32 bb; bb.lo = p[0]; bb.hi = p[1];
        v16bf bfrag = __builtin_bit_cast(v16bf, bb);
        acc[nt] = __builtin_amdgcn_wmma_f32_16x16x32_bf16(
            false, afrag, false, bfrag, (short)0, acc[nt], false, false);
      }
    }

    // X1 = sigmoid(acc + b1) -> bf16 LDS tile 16x96 (cols 80..95 zero pad)
    {
      uint4 z4 = {0, 0, 0, 0};
      *(uint4*)&x1[lrow * 96 + 80 + 8 * half] = z4;
    }
    #pragma unroll
    for (int nt = 0; nt < 5; ++nt) {
      const int col = 16 * nt + lrow;          // C-layout: lane holds col N
      const float bias = b1s[col];
      #pragma unroll
      for (int r = 0; r < 8; ++r) {
        const int m = r + 8 * half;            // C-layout row
        float v = sigmoidf_(acc[nt][r] + bias);
        x1[m * 96 + col] = __builtin_bit_cast(unsigned short, (__bf16)v);
      }
    }
    asm volatile("s_wait_dscnt 0x0" ::: "memory");

    // ---- GEMM2: X1(16x96 bf16) @ W2p(96x48 bf16) -> 16x48 f32 ----
    v8f acc2[3];
    #pragma unroll
    for (int nt = 0; nt < 3; ++nt) acc2[nt] = (v8f){0,0,0,0,0,0,0,0};
    #pragma unroll
    for (int ki = 0; ki < 3; ++ki) {
      BF32 ab;
      ab.lo = *(const uint4*)&x1[lrow * 96 + 32 * ki + 8 * half];
      ab.hi = *(const uint4*)&x1[lrow * 96 + 32 * ki + 16 + 8 * half];
      v16bf afrag = __builtin_bit_cast(v16bf, ab);
      #pragma unroll
      for (int nt = 0; nt < 3; ++nt) {
        const uint4* p = W2p + (((ki * 3 + nt) * 32 + lane) << 1);
        BF32 bb; bb.lo = p[0]; bb.hi = p[1];
        v16bf bfrag = __builtin_bit_cast(v16bf, bb);
        acc2[nt] = __builtin_amdgcn_wmma_f32_16x16x32_bf16(
            false, afrag, false, bfrag, (short)0, acc2[nt], false, false);
      }
    }

    // X2 = sigmoid(acc2 + b2) -> f32 LDS (16x48, cols>=40 dropped)
    #pragma unroll
    for (int nt = 0; nt < 3; ++nt) {
      const int col = 16 * nt + lrow;
      const float bias = b2s[col];
      #pragma unroll
      for (int r = 0; r < 8; ++r) {
        float v = sigmoidf_(acc2[nt][r] + bias);
        if (col < 40) xf[(r + 8 * half) * 48 + col] = v;
      }
    }
    asm volatile("s_wait_dscnt 0x0" ::: "memory");

    // ---- GEMM3: 40-wide dot per row -> raw score ----
    if (half == 0) {
      float s = b3s_sh;
      #pragma unroll 8
      for (int c = 0; c < 40; ++c) s += xf[lrow * 48 + c] * w3s[c];
      sc[wave][t0 + lrow] = s;
    }
  }

  asm volatile("s_wait_dscnt 0x0" ::: "memory");

  // ---- masked softmax over t for n = wave ----
  {
    const int L = len_sh;
    float vals[7];
    float m = -3.0e38f;
    #pragma unroll
    for (int i = 0; i < 7; ++i) {
      const int t = lane + 32 * i;
      float s = NEGC;
      if (t < T_) {
        s = sc[wave][t];
        if (t >= L) s = NEGC;                  // mask BEFORE scaling (ref)
      }
      s *= 0.125f;                             // 1/sqrt(H)
      vals[i] = s;
      m = fmaxf(m, s);
    }
    #pragma unroll
    for (int off = 16; off >= 1; off >>= 1) m = fmaxf(m, __shfl_xor(m, off, 32));
    float sum = 0.0f;
    float ev[7];
    #pragma unroll
    for (int i = 0; i < 7; ++i) { ev[i] = __expf(vals[i] - m); sum += ev[i]; }
    #pragma unroll
    for (int off = 16; off >= 1; off >>= 1) sum += __shfl_xor(sum, off, 32);
    const float inv = 1.0f / sum;
    #pragma unroll
    for (int i = 0; i < 7; ++i) {
      const int t = lane + 32 * i;
      if (t < 208) sc[wave][t] = ev[i] * inv;  // t>=200 -> ~0, unused anyway
    }
  }
  __syncthreads();

  // ---- out[b, nb..nb+7, :] = w(8x200) @ keys(200x64), exact f32 WMMA ----
  if (wave < 4) {                              // wave w owns h-tile 16w..16w+15
    const int hc = 16 * wave + lrow;
    const float* kb = keys + (size_t)b * T_ * H_;
    v8f oacc = (v8f){0,0,0,0,0,0,0,0};
    for (int kt = 0; kt < 50; ++kt) {
      const int tA = 4 * kt + 2 * half;        // f32 A/B frag: K = 2*half + j
      v2f a, bv;
      a[0] = (lrow < 8) ? sc[lrow][tA]     : 0.0f;   // rows 8..15 zero-padded
      a[1] = (lrow < 8) ? sc[lrow][tA + 1] : 0.0f;
      bv[0] = kb[tA * H_ + hc];
      bv[1] = kb[(tA + 1) * H_ + hc];
      oacc = __builtin_amdgcn_wmma_f32_16x16x4_f32(
          false, a, false, bv, (short)0, oacc, false, false);
    }
    if (half == 0) {                           // C rows 0..7 = valid n rows
      #pragma unroll
      for (int r = 0; r < 8; ++r)
        out[((size_t)b * N_ + nb + r) * H_ + hc] = oacc[r];
    }
  }
}

extern "C" void kernel_launch(void* const* d_in, const int* in_sizes, int n_in,
                              void* d_out, int out_size, void* d_ws, size_t ws_size,
                              hipStream_t stream) {
  (void)in_sizes; (void)n_in; (void)out_size; (void)ws_size;
  const float* queries = (const float*)d_in[0];
  const float* keys    = (const float*)d_in[1];
  const int*   klen    = (const int*)d_in[2];
  const float* W1 = (const float*)d_in[3];
  const float* b1 = (const float*)d_in[4];
  const float* W2 = (const float*)d_in[5];
  const float* b2 = (const float*)d_in[6];
  const float* W3 = (const float*)d_in[7];
  const float* b3 = (const float*)d_in[8];
  float* out = (float*)d_out;
  unsigned int* ws = (unsigned int*)d_ws;      // needs 12544*4 = ~50KB

  pack_weights_kernel<<<49, 256, 0, stream>>>(W1, W2, ws);
  din_attention_kernel<<<B_ * 2, 256, 0, stream>>>(
      queries, keys, klen, b1, b2, W3, b3, ws, out);
}
